// VideoCompressor_84181359001717
// MI455X (gfx1250) — matrix operand order repaired and checked by reading, not compiled, not run on hardware
//
#include <hip/hip_runtime.h>
#include <hip/hip_bf16.h>

// ---------- problem constants ----------
#define NTOK   2048          // BT*S
#define LTOT   25
#define KD     4096          // D
#define CC     1024          // C
#define NHEAD  8
#define HD     128
#define NKV    6             // J-1
#define NBATCH 8             // 7 LN'd layers + raw layer-24 for vin

typedef __attribute__((ext_vector_type(16))) __bf16 v16bf;
typedef __attribute__((ext_vector_type(8)))  __bf16 v8bf;
typedef __attribute__((ext_vector_type(8)))  float  v8f;

// ---------- workspace layout (bytes) ----------
static constexpr size_t OFF_W   = 0;                                    // bf16 [8][CC][KD]
static constexpr size_t SZ_W    = (size_t)NBATCH * CC * KD * 2;         // 64 MB
static constexpr size_t OFF_OW  = OFF_W + SZ_W;                         // bf16 [CC][CC]
static constexpr size_t SZ_OW   = (size_t)CC * CC * 2;                  // 2 MB
static constexpr size_t OFF_A   = OFF_OW + SZ_OW;                       // bf16 [8][NTOK][KD]
static constexpr size_t SZ_A    = (size_t)NBATCH * NTOK * KD * 2;       // 128 MB
static constexpr size_t OFF_VO  = OFF_A + SZ_A;                         // f32  [8][NTOK][CC]
static constexpr size_t SZ_VO   = (size_t)NBATCH * NTOK * CC * 4;       // 64 MB
static constexpr size_t OFF_F   = OFF_VO + SZ_VO;                       // bf16 [NTOK][CC]
static constexpr size_t SZ_F    = (size_t)NTOK * CC * 2;                // 4 MB
static constexpr size_t OFF_VF  = OFF_F + SZ_F;                         // f32  [NTOK][CC]
static constexpr size_t SZ_VF   = (size_t)NTOK * CC * 4;                // 8 MB
static constexpr size_t OFF_COS = OFF_VF + SZ_VF;                       // f32  [NTOK]

// =====================================================================
// f32 -> bf16 bulk convert (4 elems / thread, n multiple of 1024)
// =====================================================================
__global__ void cvt_bf16(const float* __restrict__ in, __bf16* __restrict__ out,
                         long long n) {
    long long i = ((long long)blockIdx.x * 256 + threadIdx.x) * 4;
    if (i + 3 < n) {
        float4 f = *(const float4*)(in + i);
        out[i + 0] = (__bf16)f.x;
        out[i + 1] = (__bf16)f.y;
        out[i + 2] = (__bf16)f.z;
        out[i + 3] = (__bf16)f.w;
    }
}

// =====================================================================
// LayerNorm + bf16 pack.  grid = (NTOK, 8), block = 256.
// slot j<7: layer 24-4j with LN(gamma[j],beta[j]); slot 7: raw layer 24.
// =====================================================================
__global__ void ln_pack(const float* __restrict__ feats,   // [NTOK][25][KD]
                        const float* __restrict__ gamma,   // [7][KD]
                        const float* __restrict__ beta,    // [7][KD]
                        __bf16* __restrict__ A) {          // [8][NTOK][KD]
    const int t   = blockIdx.x;
    const int j   = blockIdx.y;
    const int tid = threadIdx.x;
    const int layer = (j < 7) ? (24 - 4 * j) : 24;
    const float* x = feats + ((size_t)t * LTOT + layer) * KD;
    __bf16* out    = A + ((size_t)j * NTOK + t) * KD;

    float v[16];
    float s = 0.f, s2 = 0.f;
#pragma unroll
    for (int i = 0; i < 16; ++i) {
        float f = x[tid + 256 * i];
        v[i] = f; s += f; s2 += f * f;
    }
    __shared__ float ss[256], sq[256];
    ss[tid] = s; sq[tid] = s2;
    __syncthreads();
    for (int o = 128; o > 0; o >>= 1) {
        if (tid < o) { ss[tid] += ss[tid + o]; sq[tid] += sq[tid + o]; }
        __syncthreads();
    }
    const float mean = ss[0] * (1.f / KD);
    const float var  = sq[0] * (1.f / KD) - mean * mean;
    const float rstd = rsqrtf(var + 1e-5f);

    if (j < 7) {
        const float* g = gamma + (size_t)j * KD;
        const float* b = beta  + (size_t)j * KD;
#pragma unroll
        for (int i = 0; i < 16; ++i) {
            int d = tid + 256 * i;
            out[d] = (__bf16)(((v[i] - mean) * rstd) * g[d] + b[d]);
        }
    } else {
#pragma unroll
        for (int i = 0; i < 16; ++i) out[tid + 256 * i] = (__bf16)v[i];
    }
}

// =====================================================================
// WMMA helpers (bf16 16x16x32, f32 accum) — per-ISA VGPR layouts
// =====================================================================
__device__ __forceinline__ v16bf frag_join(v8bf lo, v8bf hi) {
    v16bf r;
#pragma unroll
    for (int i = 0; i < 8; ++i) { r[i] = lo[i]; r[i + 8] = hi[i]; }
    return r;
}
// A 16x32 bf16: lane = M (mod 16); half-wave 0 -> K {0..7,16..23}, half 1 -> +8
__device__ __forceinline__ v16bf load_a_frag(const __bf16* p) {
    return frag_join(*(const v8bf*)p, *(const v8bf*)(p + 16));
}
// B 32x16 bf16: lane = N (mod 16); half-wave 0 -> K {0..7,16..23}, half 1 -> +8
__device__ __forceinline__ v16bf load_b_frag(const __bf16* p) {
    return frag_join(*(const v8bf*)p, *(const v8bf*)(p + 16));
}

// ---------------------------------------------------------------------
// One wave computes a 64x64 output tile: 4x4 WMMA accumulators.
// Per 32-K step: 8 fragment loads feed 16 WMMAs (4x reuse per operand).
// A[M][K], B/W[N][K], both K-contiguous row-major.
// ---------------------------------------------------------------------
__device__ __forceinline__ void wmma_tile_64x64(
        const __bf16* __restrict__ Abase,   // row m0, col 0
        const __bf16* __restrict__ Bbase,   // row n0, col 0
        int K, int lane, v8f acc[4][4]) {
    const int half = lane >> 4;
    const int l15  = lane & 15;
    const __bf16* pa[4];
    const __bf16* pb[4];
#pragma unroll
    for (int i = 0; i < 4; ++i) {
        pa[i] = Abase + (size_t)(l15 + 16 * i) * K + half * 8;
        pb[i] = Bbase + (size_t)(l15 + 16 * i) * K + half * 8;
    }
    for (int k0 = 0; k0 < K; k0 += 32) {
        v16bf fa[4], fb[4];
#pragma unroll
        for (int i = 0; i < 4; ++i) fa[i] = load_a_frag(pa[i] + k0);
#pragma unroll
        for (int i = 0; i < 4; ++i) fb[i] = load_b_frag(pb[i] + k0);
        if (k0 + 32 < K) {            // pull next K-slab toward the WGP
            __builtin_prefetch(pa[0] + k0 + 32, 0, 3);
            __builtin_prefetch(pb[0] + k0 + 32, 0, 3);
        }
#pragma unroll
        for (int mi = 0; mi < 4; ++mi)
#pragma unroll
            for (int ni = 0; ni < 4; ++ni)
                acc[mi][ni] = __builtin_amdgcn_wmma_f32_16x16x32_bf16(
                    false, fa[mi], false, fb[ni], (short)0, acc[mi][ni],
                    false, false);
    }
}

// C/D 16x16 f32: VGPR r -> M = r + 8*half ; lane&15 -> N
__device__ __forceinline__ void store_tile(float* __restrict__ O, int ldo,
                                           const v8f& acc, int mbase, int nbase,
                                           int lane) {
    const int m = mbase + (lane >> 4) * 8;
    const int n = nbase + (lane & 15);
#pragma unroll
    for (int r = 0; r < 8; ++r) O[(size_t)(m + r) * ldo + n] = acc[r];
}

// =====================================================================
// Batched GEMM: VO[b][t][c] = A[b][t][:] . W[b][c][:]
// block = 256 (8 waves: 2 along M x 4 along N), block tile 128x256
// grid (2048/128, 1024/256, 8) = (16, 4, 8)
// =====================================================================
__global__ void gemm_vo(const __bf16* __restrict__ A,   // [8][NTOK][KD]
                        const __bf16* __restrict__ W,   // [8][CC][KD]
                        float* __restrict__ VO) {       // [8][NTOK][CC]
    const int b    = blockIdx.z;
    const int lane = threadIdx.x & 31;
    const int wave = threadIdx.x >> 5;
    const int m0 = blockIdx.x * 128 + (wave & 1) * 64;
    const int n0 = blockIdx.y * 256 + (wave >> 1) * 64;

    v8f acc[4][4] = {};
    wmma_tile_64x64(A + (size_t)b * NTOK * KD + (size_t)m0 * KD,
                    W + (size_t)b * CC   * KD + (size_t)n0 * KD,
                    KD, lane, acc);

    float* Ob = VO + (size_t)b * NTOK * CC;
#pragma unroll
    for (int mi = 0; mi < 4; ++mi)
#pragma unroll
        for (int ni = 0; ni < 4; ++ni)
            store_tile(Ob, CC, acc[mi][ni], m0 + 16 * mi, n0 + 16 * ni, lane);
}

// =====================================================================
// Per-token attention fuse.  grid = NTOK, block = 256 (wave == head).
// q = VO[0], kv = VO[1..6]; softmax over 6 keys; fuse -> bf16
// =====================================================================
__global__ void attn_fuse(const float* __restrict__ VO,   // [8][NTOK][CC]
                          __bf16* __restrict__ F) {       // [NTOK][CC]
    const int t    = blockIdx.x;
    const int h    = threadIdx.x >> 5;
    const int lane = threadIdx.x & 31;
    const float SCALE = 0.03125f;                         // 1024^-0.5
    const size_t tok = (size_t)t * CC;
    const int dbase = h * HD + lane * 4;

    float q[4], kv[NKV][4];
#pragma unroll
    for (int i = 0; i < 4; ++i) q[i] = VO[tok + dbase + i] * SCALE;
#pragma unroll
    for (int k = 0; k < NKV; ++k)
#pragma unroll
        for (int i = 0; i < 4; ++i)
            kv[k][i] = VO[(size_t)(k + 1) * NTOK * CC + tok + dbase + i];

    float sc[NKV];
#pragma unroll
    for (int k = 0; k < NKV; ++k) {
        float p = q[0] * kv[k][0] + q[1] * kv[k][1] +
                  q[2] * kv[k][2] + q[3] * kv[k][3];
        for (int off = 16; off >= 1; off >>= 1) p += __shfl_xor(p, off, 32);
        sc[k] = p;
    }
    float mx = sc[0];
#pragma unroll
    for (int k = 1; k < NKV; ++k) mx = fmaxf(mx, sc[k]);
    float den = 0.f;
#pragma unroll
    for (int k = 0; k < NKV; ++k) { sc[k] = __expf(sc[k] - mx); den += sc[k]; }
    const float inv = 1.f / den;

    float out[4] = {0.f, 0.f, 0.f, 0.f};
#pragma unroll
    for (int k = 0; k < NKV; ++k)
#pragma unroll
        for (int i = 0; i < 4; ++i) out[i] += sc[k] * kv[k][i];
#pragma unroll
    for (int i = 0; i < 4; ++i) F[tok + dbase + i] = (__bf16)(out[i] * inv);
}

// =====================================================================
// Output GEMM + epilogue: VF = fuse @ OWᵀ + vin + input_b + output_b
// same tiling as gemm_vo; grid (16, 4, 1); K = 1024
// =====================================================================
__global__ void gemm_vf(const __bf16* __restrict__ F,    // [NTOK][CC]
                        const __bf16* __restrict__ OW,   // [CC][CC]
                        const float*  __restrict__ VIN,  // VO slot 7: [NTOK][CC]
                        const float*  __restrict__ inb,  // [CC]
                        const float*  __restrict__ outb, // [CC]
                        float* __restrict__ VF) {        // [NTOK][CC]
    const int lane = threadIdx.x & 31;
    const int wave = threadIdx.x >> 5;
    const int m0 = blockIdx.x * 128 + (wave & 1) * 64;
    const int n0 = blockIdx.y * 256 + (wave >> 1) * 64;

    v8f acc[4][4] = {};
    wmma_tile_64x64(F + (size_t)m0 * CC, OW + (size_t)n0 * CC,
                    CC, lane, acc);

    const int half = lane >> 4, l15 = lane & 15;
#pragma unroll
    for (int mi = 0; mi < 4; ++mi) {
#pragma unroll
        for (int ni = 0; ni < 4; ++ni) {
            const int mb = m0 + 16 * mi + half * 8;
            const int n  = n0 + 16 * ni + l15;
            const float bias = inb[n] + outb[n];
#pragma unroll
            for (int r = 0; r < 8; ++r) {
                const size_t idx = (size_t)(mb + r) * CC + n;
                VF[idx] = acc[mi][ni][r] + VIN[idx] + bias;
            }
        }
    }
}

// =====================================================================
// Per-token cosine vs teacher; then scalar 1 - mean
// =====================================================================
__global__ void cos_kernel(const float* __restrict__ VF,
                           const float* __restrict__ T,
                           float* __restrict__ cosv) {
    const int t   = blockIdx.x;
    const int tid = threadIdx.x;
    float d = 0.f, a = 0.f, b = 0.f;
    for (int i = tid; i < CC; i += 256) {
        float x = VF[(size_t)t * CC + i];
        float y = T [(size_t)t * CC + i];
        d += x * y; a += x * x; b += y * y;
    }
    __shared__ float sd[256], sa[256], sb[256];
    sd[tid] = d; sa[tid] = a; sb[tid] = b;
    __syncthreads();
    for (int o = 128; o > 0; o >>= 1) {
        if (tid < o) { sd[tid] += sd[tid + o]; sa[tid] += sa[tid + o]; sb[tid] += sb[tid + o]; }
        __syncthreads();
    }
    if (tid == 0) {
        float na = fmaxf(sqrtf(sa[0]), 1e-8f);
        float nb = fmaxf(sqrtf(sb[0]), 1e-8f);
        cosv[t] = sd[0] / (na * nb);
    }
}

__global__ void final_kernel(const float* __restrict__ cosv, float* __restrict__ out) {
    const int tid = threadIdx.x;
    float acc = 0.f;
    for (int i = tid; i < NTOK; i += 256) acc += cosv[i];
    __shared__ float s[256];
    s[tid] = acc;
    __syncthreads();
    for (int o = 128; o > 0; o >>= 1) {
        if (tid < o) s[tid] += s[tid + o];
        __syncthreads();
    }
    if (tid == 0) out[0] = 1.f - s[0] * (1.f / NTOK);
}

// =====================================================================
extern "C" void kernel_launch(void* const* d_in, const int* in_sizes, int n_in,
                              void* d_out, int out_size, void* d_ws, size_t ws_size,
                              hipStream_t stream) {
    const float* feats    = (const float*)d_in[0];   // [8][256][25][4096]
    const float* teacher  = (const float*)d_in[1];   // [8][256][1024]
    const float* gamma    = (const float*)d_in[2];   // [7][4096]
    const float* beta     = (const float*)d_in[3];   // [7][4096]
    const float* norm_w   = (const float*)d_in[4];   // [7][1024][4096]
    const float* input_w  = (const float*)d_in[5];   // [1024][4096]
    const float* input_b  = (const float*)d_in[6];   // [1024]
    const float* output_w = (const float*)d_in[7];   // [1024][1024]
    const float* output_b = (const float*)d_in[8];   // [1024]
    (void)in_sizes; (void)n_in; (void)out_size; (void)ws_size;

    char* ws = (char*)d_ws;
    __bf16* Wbf  = (__bf16*)(ws + OFF_W);
    __bf16* OWbf = (__bf16*)(ws + OFF_OW);
    __bf16* Abf  = (__bf16*)(ws + OFF_A);
    float*  VO   = (float*) (ws + OFF_VO);
    __bf16* Fbf  = (__bf16*)(ws + OFF_F);
    float*  VF   = (float*) (ws + OFF_VF);
    float*  COSV = (float*) (ws + OFF_COS);

    // 1) weights -> bf16 (norm_w as batches 0..6, input_w as batch 7, output_w)
    {
        long long n1 = (long long)7 * CC * KD;
        cvt_bf16<<<dim3((unsigned)(n1 / 1024)), 256, 0, stream>>>(norm_w, Wbf, n1);
        long long n2 = (long long)CC * KD;
        cvt_bf16<<<dim3((unsigned)(n2 / 1024)), 256, 0, stream>>>(
            input_w, Wbf + (size_t)7 * CC * KD, n2);
        long long n3 = (long long)CC * CC;
        cvt_bf16<<<dim3((unsigned)(n3 / 1024)), 256, 0, stream>>>(output_w, OWbf, n3);
    }
    // 2) LayerNorm + pack activations (slot 7 = raw layer 24)
    ln_pack<<<dim3(NTOK, NBATCH), 256, 0, stream>>>(feats, gamma, beta, Abf);
    // 3) big batched WMMA GEMM: 8 x (2048 x 1024 x 4096)
    gemm_vo<<<dim3(16, 4, NBATCH), 256, 0, stream>>>(Abf, Wbf, VO);
    // 4) attention fuse
    attn_fuse<<<dim3(NTOK), 256, 0, stream>>>(VO, Fbf);
    // 5) output WMMA GEMM + bias/vin epilogue
    gemm_vf<<<dim3(16, 4, 1), 256, 0, stream>>>(
        Fbf, OWbf, VO + (size_t)7 * NTOK * CC, input_b, output_b, VF);
    // 6) cosine loss
    cos_kernel<<<dim3(NTOK), 256, 0, stream>>>(VF, teacher, COSV);
    final_kernel<<<1, 256, 0, stream>>>(COSV, (float*)d_out);
}
